// PolyRPEAttention_13743895347458
// MI455X (gfx1250) — matrix operand order repaired and checked by reading, not compiled, not run on hardware
//
#include <hip/hip_runtime.h>
#include <hip/hip_bf16.h>

// ---------------------------------------------------------------------------
// PolyRPE attention for MI455X (gfx1250, wave32, WMMA).
// All matmuls on V_WMMA_F32_16X16X32_BF16 (f32 accumulate); softmax/rpe/bias
// in fp32. Compute-bound (~60 GFLOP vs ~60MB traffic) => matrix cores.
//
// Round-3 changes (driven by round-2 disassembly):
//  * Attention chunk loop software-pipelined: next chunk's K fragments and
//    rpe C tiles are loaded into registers a full iteration ahead, so their
//    s_wait_loadcnt no longer sits directly in front of each WMMA.
//  * Current-chunk V fragments load right after the S WMMAs issue: softmax
//    VALU hides their latency and fills WMMA->VALU hazard slots (was v_nops).
//  * GEMM mainloop register-stages the next K-slice's global loads between
//    the barrier and the WMMAs (global latency overlapped with compute).
// ---------------------------------------------------------------------------

typedef __attribute__((ext_vector_type(16))) __bf16 v16bf;
typedef __attribute__((ext_vector_type(8)))  __bf16 v8bf;
typedef __attribute__((ext_vector_type(8)))  float  v8f;

#define DIM     768
#define NHEAD   12
#define HDIM    64
#define NTOK    577          // 24*24 + 1 (CLS)
#define NPAD    592          // 37*16
#define HW      576
#define BATCH   16
#define MROWS   (BATCH * NTOK)   // 9232
#define QKV_N   (3 * DIM)        // 2304

__device__ __forceinline__ v8f wmma_bf16f32(v16bf a, v16bf b, v8f c) {
  return __builtin_amdgcn_wmma_f32_16x16x32_bf16(false, a, false, b, (short)0, c,
                                                 false, false);
}

__device__ __forceinline__ v16bf frag16(const __bf16* p0, const __bf16* p1) {
  v8bf lo = *reinterpret_cast<const v8bf*>(p0);   // elements 0..7
  v8bf hi = *reinterpret_cast<const v8bf*>(p1);   // elements 8..15
  v16bf r;
#pragma unroll
  for (int i = 0; i < 8; ++i) { r[i] = lo[i]; r[i + 8] = hi[i]; }
  return r;
}

__device__ __forceinline__ v16bf zfrag16() {
  v16bf z;
#pragma unroll
  for (int i = 0; i < 16; ++i) z[i] = (__bf16)0.0f;
  return z;
}

// 16-lane xor-butterfly max via ds_swizzle (group-of-32 mode:
// offset = xor<<10 | or<<5 | and, and=0x1f, or=0).
template <int OFF>
__device__ __forceinline__ float max_swz(float v) {
  const float o =
      __int_as_float(__builtin_amdgcn_ds_swizzle(__float_as_int(v), OFF));
  return fmaxf(v, o);
}

// ---------------------------------------------------------------------------
// Kernel 1: rpe_padded[h][x][y] = sum_i coeffs[h][i] * powers[i][x-1][y-1]
// ---------------------------------------------------------------------------
__global__ __launch_bounds__(256) void rpe_build(const float* __restrict__ coeffs,
                                                 const float* __restrict__ powers,
                                                 float* __restrict__ rpe) {
  const int idx = blockIdx.x * 256 + threadIdx.x;
  const int total = NHEAD * NTOK * NTOK;
  if (idx >= total) return;
  const int h   = idx / (NTOK * NTOK);
  const int rem = idx - h * (NTOK * NTOK);
  const int xx  = rem / NTOK;
  const int yy  = rem - xx * NTOK;
  float v = 0.0f;
  if (xx > 0 && yy > 0) {
    const int p = (xx - 1) * HW + (yy - 1);
    const float* c = coeffs + h * 4;
    v = c[0] * powers[p]
      + c[1] * powers[1 * HW * HW + p]
      + c[2] * powers[2 * HW * HW + p]
      + c[3] * powers[3 * HW * HW + p];
  }
  rpe[idx] = v;
}

// ---------------------------------------------------------------------------
// Kernel 2/4: bf16 WMMA GEMM, C = A(fp32) * B(fp32), fp32 accumulate.
//   Block 64x128, 8 waves (4Mx2N), each wave 16x64 (4 independent acc tiles).
//   Register-staged pipeline: global loads for slice k0+32 issue between the
//   first barrier and the WMMAs of slice k0.
// ---------------------------------------------------------------------------
__device__ __forceinline__ void gemm_stage_load(
    const float* __restrict__ A, const float* __restrict__ Bw, int Ntot,
    int m0, int n0, int arow, int acol, int brow, int bcol, int k0,
    float* av, float* bv) {
  const int grow = m0 + arow;
  if (grow < MROWS) {
    const float4* pa =
        reinterpret_cast<const float4*>(A + (size_t)grow * DIM + k0 + acol);
    float4 a0 = pa[0], a1 = pa[1];
    av[0] = a0.x; av[1] = a0.y; av[2] = a0.z; av[3] = a0.w;
    av[4] = a1.x; av[5] = a1.y; av[6] = a1.z; av[7] = a1.w;
  } else {
#pragma unroll
    for (int i = 0; i < 8; ++i) av[i] = 0.0f;
  }
  const float4* pb4 = reinterpret_cast<const float4*>(
      Bw + (size_t)(k0 + brow) * Ntot + n0 + bcol);
#pragma unroll
  for (int q = 0; q < 4; ++q) {
    float4 v = pb4[q];
    bv[q * 4 + 0] = v.x; bv[q * 4 + 1] = v.y;
    bv[q * 4 + 2] = v.z; bv[q * 4 + 3] = v.w;
  }
}

template <int MODE>   // 0 = qkv scatter, 1 = proj + bias
__global__ __launch_bounds__(256) void gemm_bf16_wmma(
    const float* __restrict__ A, const float* __restrict__ Bw, const int Ntot,
    const float* __restrict__ bias, float* __restrict__ Cout,
    __bf16* __restrict__ Qbf, __bf16* __restrict__ Kbf, __bf16* __restrict__ Vt) {
  __shared__ __bf16 lA[64 * 40];    // [m][k], 64x32 used
  __shared__ __bf16 lB[128 * 40];   // [n][k], 128x32 used (transposed)

  const int t    = threadIdx.x;
  const int lane = t & 31;
  const int wid  = t >> 5;
  const int half = lane >> 4;
  const int n16  = lane & 15;
  const int wm   = wid & 3;
  const int wn   = wid >> 2;
  const int m0   = blockIdx.x * 64;
  const int n0   = blockIdx.y * 128;

  v8f acc[4];
#pragma unroll
  for (int i = 0; i < 4; ++i)
#pragma unroll
    for (int r = 0; r < 8; ++r) acc[i][r] = 0.0f;

  const int arow = t >> 2;         // 0..63
  const int acol = (t & 3) * 8;    // 0,8,16,24
  const int brow = t >> 3;         // 0..31
  const int bcol = (t & 7) * 16;   // 0..112

  float avr[8], bvr[16];
  gemm_stage_load(A, Bw, Ntot, m0, n0, arow, acol, brow, bcol, 0, avr, bvr);

  for (int k0 = 0; k0 < DIM; k0 += 32) {
    // ---- commit staged registers to LDS (fp32 -> bf16) ----
#pragma unroll
    for (int i = 0; i < 8; ++i) lA[arow * 40 + acol + i] = (__bf16)avr[i];
#pragma unroll
    for (int i = 0; i < 16; ++i) lB[(bcol + i) * 40 + brow] = (__bf16)bvr[i];
    __syncthreads();

    // ---- issue next slice's global loads; latency hidden by WMMAs ----
    if (k0 + 32 < DIM)
      gemm_stage_load(A, Bw, Ntot, m0, n0, arow, acol, brow, bcol, k0 + 32,
                      avr, bvr);
    if (k0 + 64 < DIM) {   // keep L2 warm two slices ahead
      if (m0 + arow < MROWS)
        __builtin_prefetch(A + (size_t)(m0 + arow) * DIM + (k0 + 64) + acol, 0, 1);
      __builtin_prefetch(Bw + (size_t)(k0 + 64 + brow) * Ntot + n0 + bcol, 0, 1);
    }

    // ---- WMMAs for current slice ----
    const __bf16* pa0 = &lA[(wm * 16 + n16) * 40 + half * 8];
    const v16bf af = frag16(pa0, pa0 + 16);
#pragma unroll
    for (int nt = 0; nt < 4; ++nt) {
      const __bf16* pb0 = &lB[(wn * 64 + nt * 16 + n16) * 40 + half * 8];
      const v16bf bf_ = frag16(pb0, pb0 + 16);
      acc[nt] = wmma_bf16f32(af, bf_, acc[nt]);   // 4 independent chains
    }
    __syncthreads();
  }

  // ---- store: C/D layout lane holds (M = r + 8*half, N = n16) ----
#pragma unroll
  for (int nt = 0; nt < 4; ++nt) {
#pragma unroll
    for (int r = 0; r < 8; ++r) {
      const int row = m0 + wm * 16 + r + 8 * half;
      const int col = n0 + wn * 64 + nt * 16 + n16;
      if (row < MROWS) {
        const float v = acc[nt][r];
        if (MODE == 1) {
          Cout[(size_t)row * DIM + col] = v + bias[col];
        } else {
          const int b   = row / NTOK;
          const int n   = row - b * NTOK;
          const int sec = col / DIM;          // 0=Q 1=K 2=V (uniform per block)
          const int c   = col - sec * DIM;
          const int h   = c >> 6;
          const int dd  = c & 63;
          const size_t bh = (size_t)(b * NHEAD + h);
          if (sec == 0)
            Qbf[(bh * NPAD + n) * HDIM + dd] = (__bf16)(v * 0.125f);  // d^-0.5
          else if (sec == 1)
            Kbf[(bh * NPAD + n) * HDIM + dd] = (__bf16)v;
          else
            Vt[(bh * HDIM + dd) * NPAD + n] = (__bf16)v;   // transposed
        }
      }
    }
  }
}

// ---------------------------------------------------------------------------
// Kernel 3: flash attention with polynomial RPE (software-pipelined).
// ---------------------------------------------------------------------------
__device__ __forceinline__ void load_k2(const __bf16* __restrict__ kbase,
                                        int kcol, int half,
                                        v16bf& f0, v16bf& f1) {
  if (kcol < NPAD) {
    const __bf16* p = kbase + (size_t)kcol * HDIM + half * 8;
    f0 = frag16(p, p + 16);           // dd  0..31
    f1 = frag16(p + 32, p + 48);      // dd 32..63
  } else {
    f0 = zfrag16();
    f1 = zfrag16();
  }
}

__device__ __forceinline__ void load_cc2(const float* __restrict__ rbase,
                                         int qt, int half, int n16, int j0,
                                         v8f& c0, v8f& c1) {
  const int kcol0 = j0 + n16;
  const int kcol1 = j0 + 16 + n16;
#pragma unroll
  for (int r = 0; r < 8; ++r) {
    const int qr = qt * 16 + r + 8 * half;
    const bool qok = qr < NTOK;
    c0[r] = (qok && kcol0 < NTOK) ? rbase[(size_t)qr * NTOK + kcol0] : 0.0f;
    c1[r] = (qok && kcol1 < NTOK) ? rbase[(size_t)qr * NTOK + kcol1] : 0.0f;
  }
}

__global__ __launch_bounds__(128) void attn_poly_wmma(
    const __bf16* __restrict__ Q, const __bf16* __restrict__ Kb,
    const __bf16* __restrict__ Vt, const float* __restrict__ rpe,
    float* __restrict__ ctx) {
  __shared__ __bf16 lp[4][16 * 40];   // per-wave 16x32 P tile (stride 40)

  const int lane = threadIdx.x & 31;
  const int wid  = threadIdx.x >> 5;
  const int half = lane >> 4;
  const int n16  = lane & 15;
  const int qt   = blockIdx.x * 4 + wid;
  if (qt >= (NPAD / 16)) return;      // wave-uniform; no block barriers used
  const int bh = blockIdx.y;
  const int b  = bh / NHEAD;
  const int h  = bh - b * NHEAD;

  const __bf16* qbase = Q  + (size_t)bh * NPAD * HDIM;
  const __bf16* kbase = Kb + (size_t)bh * NPAD * HDIM;
  const __bf16* vbase = Vt + (size_t)bh * HDIM * NPAD;
  const float*  rbase = rpe + (size_t)h * NTOK * NTOK;

  // Q fragments (A-layout) for dd = [0,32) and [32,64)
  v16bf aq[2];
#pragma unroll
  for (int kk = 0; kk < 2; ++kk) {
    const __bf16* p =
        qbase + (size_t)(qt * 16 + n16) * HDIM + kk * 32 + half * 8;
    aq[kk] = frag16(p, p + 16);
  }

  // all-ones B fragment for row-sum WMMA
  v16bf bones;
#pragma unroll
  for (int i = 0; i < 16; ++i) bones[i] = (__bf16)1.0f;

  v8f o[4];
#pragma unroll
  for (int i = 0; i < 4; ++i)
#pragma unroll
    for (int r = 0; r < 8; ++r) o[i][r] = 0.0f;
  float mrow[8], lrow[8];
#pragma unroll
  for (int r = 0; r < 8; ++r) { mrow[r] = -3.0e38f; lrow[r] = 0.0f; }

  // ---- pipeline prologue: chunk 0 K fragments + rpe tiles ----
  v16bf nk00, nk01, nk10, nk11;       // nk[t][kk]
  v8f ncc0, ncc1;
  load_k2(kbase, n16, half, nk00, nk01);
  load_k2(kbase, 16 + n16, half, nk10, nk11);
  load_cc2(rbase, qt, half, n16, 0, ncc0, ncc1);

  for (int c = 0; c < 19; ++c) {      // ceil(577/32) chunks of 32 k-rows
    const int j0 = c * 32;

    // rotate pipeline registers (SSA renaming; no copies in codegen)
    const v16bf bk00 = nk00, bk01 = nk01, bk10 = nk10, bk11 = nk11;
    const v8f cc0 = ncc0, cc1 = ncc1;

    // ---- prefetch next chunk's K fragments + rpe tiles ----
    if (c + 1 < 19) {
      const int jn = j0 + 32;
      load_k2(kbase, jn + n16, half, nk00, nk01);
      load_k2(kbase, jn + 16 + n16, half, nk10, nk11);
      load_cc2(rbase, qt, half, n16, jn, ncc0, ncc1);
    }

    // ---- S = q_s @ k^T + rpe : two interleaved accumulator chains ----
    v8f s0 = wmma_bf16f32(aq[0], bk00, cc0);
    v8f s1 = wmma_bf16f32(aq[0], bk10, cc1);
    s0 = wmma_bf16f32(aq[1], bk01, s0);
    s1 = wmma_bf16f32(aq[1], bk11, s1);

    // ---- V fragments for this chunk: issue now, consumed after softmax ----
    v16bf bv[4];
    {
      const int cstart = j0 + half * 16;
#pragma unroll
      for (int vt = 0; vt < 4; ++vt) {
        if (cstart < NPAD) {
          const __bf16* pv = vbase + (size_t)(vt * 16 + n16) * NPAD + cstart;
          bv[vt] = frag16(pv, pv + 8);
        } else {
          bv[vt] = zfrag16();
        }
      }
    }

    // ---- mask invalid / padded k-rows ----
    const int kcol0 = j0 + n16;
    const int kcol1 = j0 + 16 + n16;
    if (kcol0 >= NTOK) {
#pragma unroll
      for (int r = 0; r < 8; ++r) s0[r] = -3.0e38f;
    }
    if (kcol1 >= NTOK) {
#pragma unroll
      for (int r = 0; r < 8; ++r) s1[r] = -3.0e38f;
    }

    // ---- online softmax (row r lives at VGPR slot r, M = r + 8*half) ----
    float cf[8], e0v[8], e1v[8];
#pragma unroll
    for (int r = 0; r < 8; ++r) {
      float cm = fmaxf(s0[r], s1[r]);
      cm = max_swz<0x041F>(cm);   // xor 1
      cm = max_swz<0x081F>(cm);   // xor 2
      cm = max_swz<0x101F>(cm);   // xor 4
      cm = max_swz<0x201F>(cm);   // xor 8
      const float mn = fmaxf(mrow[r], cm);
      e0v[r] = __expf(s0[r] - mn);
      e1v[r] = __expf(s1[r] - mn);
      cf[r]  = __expf(mrow[r] - mn);
      mrow[r] = mn;
    }
#pragma unroll
    for (int vt = 0; vt < 4; ++vt)
#pragma unroll
      for (int r = 0; r < 8; ++r) o[vt][r] *= cf[r];

    // ---- transpose P (C layout -> A layout) through per-wave LDS ----
    __bf16* P = lp[wid];
#pragma unroll
    for (int r = 0; r < 8; ++r) {
      const int m = r + 8 * half;
      P[m * 40 + n16]      = (__bf16)e0v[r];
      P[m * 40 + 16 + n16] = (__bf16)e1v[r];
    }
    asm volatile("s_wait_dscnt 0" ::: "memory");  // in-wave LDS RAW fence
    const __bf16* pp = P + n16 * 40 + half * 16;
    const v16bf ap = frag16(pp, pp + 8);

    // ---- row-sum of P via WMMA vs all-ones B (lands in lrow's layout) ----
    v8f zc;
#pragma unroll
    for (int r = 0; r < 8; ++r) zc[r] = 0.0f;
    const v8f rs = wmma_bf16f32(ap, bones, zc);

    // ---- O += P(16x32) @ V(32x64) : 4 independent chains ----
#pragma unroll
    for (int vt = 0; vt < 4; ++vt) o[vt] = wmma_bf16f32(ap, bv[vt], o[vt]);

#pragma unroll
    for (int r = 0; r < 8; ++r) lrow[r] = lrow[r] * cf[r] + rs[r];
  }

  // ---- normalize + scatter to ctx [B, N, C] ----
#pragma unroll
  for (int vt = 0; vt < 4; ++vt)
#pragma unroll
    for (int r = 0; r < 8; ++r) {
      const int qr = qt * 16 + r + 8 * half;
      if (qr < NTOK)
        ctx[((size_t)b * NTOK + qr) * DIM + h * HDIM + vt * 16 + n16] =
            o[vt][r] / lrow[r];
    }
}

// ---------------------------------------------------------------------------
// Launch
// ---------------------------------------------------------------------------
extern "C" void kernel_launch(void* const* d_in, const int* in_sizes, int n_in,
                              void* d_out, int out_size, void* d_ws, size_t ws_size,
                              hipStream_t stream) {
  const float* x      = (const float*)d_in[0];
  const float* W_qkv  = (const float*)d_in[1];
  const float* W_proj = (const float*)d_in[2];
  const float* b_proj = (const float*)d_in[3];
  const float* coeffs = (const float*)d_in[4];
  const float* powers = (const float*)d_in[5];
  float* out = (float*)d_out;
  (void)in_sizes; (void)n_in; (void)out_size; (void)ws_size;

  char* ws = (char*)d_ws;
  size_t off = 0;
  auto carve = [&](size_t bytes) -> void* {
    void* p = ws + off;
    off += (bytes + 255) & ~(size_t)255;
    return p;
  };
  float*  rpe = (float*)carve((size_t)NHEAD * NTOK * NTOK * sizeof(float)); // ~16 MB
  __bf16* Qbf = (__bf16*)carve((size_t)BATCH * NHEAD * NPAD * HDIM * 2);    // ~14.5 MB
  __bf16* Kbf = (__bf16*)carve((size_t)BATCH * NHEAD * NPAD * HDIM * 2);
  __bf16* Vt  = (__bf16*)carve((size_t)BATCH * NHEAD * HDIM * NPAD * 2);
  float*  ctx = (float*)carve((size_t)MROWS * DIM * sizeof(float));         // ~28 MB

  {
    const int total = NHEAD * NTOK * NTOK;
    rpe_build<<<(total + 255) / 256, 256, 0, stream>>>(coeffs, powers, rpe);
  }
  gemm_bf16_wmma<0><<<dim3((MROWS + 63) / 64, QKV_N / 128), 256, 0, stream>>>(
      x, W_qkv, QKV_N, nullptr, nullptr, Qbf, Kbf, Vt);
  attn_poly_wmma<<<dim3((NPAD / 16 + 3) / 4, BATCH * NHEAD), 128, 0, stream>>>(
      Qbf, Kbf, Vt, rpe, ctx);
  gemm_bf16_wmma<1><<<dim3((MROWS + 63) / 64, DIM / 128), 256, 0, stream>>>(
      ctx, W_proj, DIM, b_proj, out, nullptr, nullptr, nullptr);
}
